// ScaledDotProductAttention_72541997629749
// MI455X (gfx1250) — matrix operand order repaired and checked by reading, not compiled, not run on hardware
//
#include <hip/hip_runtime.h>

typedef __attribute__((ext_vector_type(16))) __bf16   v16bf;
typedef __attribute__((ext_vector_type(8)))  float    v8f;
typedef __attribute__((ext_vector_type(4)))  unsigned uv4;
typedef __attribute__((ext_vector_type(4)))  float    fv4;
typedef __attribute__((ext_vector_type(4)))  int      vi4;

union ABu { unsigned u[8]; uv4 q[2]; v16bf v; };

// ---- f32 -> bf16 (RNE). Use HW packed convert when the builtin exists. ----
#if __has_builtin(__builtin_amdgcn_cvt_pk_bf16_f32)
__device__ __forceinline__ unsigned pack2(float lo, float hi) {
  auto p = __builtin_amdgcn_cvt_pk_bf16_f32(lo, hi);
  return __builtin_bit_cast(unsigned, p);
}
__device__ __forceinline__ unsigned short f2bf(float f) {
  return (unsigned short)pack2(f, f);
}
#else
__device__ __forceinline__ unsigned short f2bf(float f) {
  unsigned x = __float_as_uint(f);
  x += 0x7FFFu + ((x >> 16) & 1u);
  return (unsigned short)(x >> 16);
}
__device__ __forceinline__ unsigned pack2(float lo, float hi) {
  return (unsigned)f2bf(lo) | ((unsigned)f2bf(hi) << 16);
}
#endif

// ---- 16B global -> LDS copy: async (ASYNCcnt) when available, else sync. ----
#if __has_builtin(__builtin_amdgcn_global_load_async_to_lds_b128)
__device__ __forceinline__ void cp16(const void* g, void* l) {
  // param types are generic v4i* (per compiler diagnostic); order follows the
  // global_load_lds precedent: (global src, LDS dst, imm offset, cpol)
  __builtin_amdgcn_global_load_async_to_lds_b128((vi4*)g, (vi4*)l, 0, 0);
}
__device__ __forceinline__ void cp16_wait() {
  asm volatile("s_wait_asynccnt 0x0" ::: "memory");
}
#else
__device__ __forceinline__ void cp16(const void* g, void* l) {
  *(uv4*)l = *(const uv4*)g;
}
__device__ __forceinline__ void cp16_wait() {}
#endif

// ---------------------------------------------------------------------------
// Pass 1: OUT[row][0:128] = X[row][0:512] @ Wm[512][128], f32 in, bf16 out.
// Chunk-outer / 8 persistent accumulators: X read exactly once. W chunk
// (32x128) staged to LDS (padded rows -> conflict-free ds_load_b128).
// MODE 0: row-major [rows][128]; MODE 1: transposed [B][128][2048] (for V).
// ---------------------------------------------------------------------------
template <int MODE>
__global__ __launch_bounds__(256)
void proj_kernel(const float* __restrict__ X, const float* __restrict__ Wm,
                 unsigned short* __restrict__ out) {
  __shared__ __align__(16) unsigned wt[128][20];  // [col][k-pair], 80B stride

  const int lane = threadIdx.x & 31;
  const int wave = threadIdx.x >> 5;
  const int rowBase = blockIdx.x * 128 + wave * 16;
  const int mA = lane & 15;
  const int hi = (lane >> 4) & 1;
  const int nB = lane & 15;

  const int scol = threadIdx.x & 127;        // staging: column 0..127
  const int sj0  = (threadIdx.x >> 7) * 8;   // staging: first k-pair (0 or 8)

  const float* xrow = X + (size_t)(rowBase + mA) * 512 + hi * 8;

  v8f acc[8];
#pragma unroll
  for (int dt = 0; dt < 8; ++dt) acc[dt] = {};

  for (int c = 0; c < 16; ++c) {             // 512/32 reduction chunks
    __syncthreads();
#pragma unroll
    for (int j = 0; j < 8; ++j) {
      const int kg = c * 32 + 2 * (sj0 + j);
      wt[scol][sj0 + j] = pack2(Wm[(size_t)kg * 128 + scol],
                                Wm[(size_t)(kg + 1) * 128 + scol]);
    }
    __syncthreads();

    ABu a;
    {
      const fv4* xa = (const fv4*)(xrow + c * 32);
      const fv4 f0 = xa[0], f1 = xa[1], f2 = xa[4], f3 = xa[5];  // +16 floats
      a.u[0] = pack2(f0.x, f0.y); a.u[1] = pack2(f0.z, f0.w);
      a.u[2] = pack2(f1.x, f1.y); a.u[3] = pack2(f1.z, f1.w);
      a.u[4] = pack2(f2.x, f2.y); a.u[5] = pack2(f2.z, f2.w);
      a.u[6] = pack2(f3.x, f3.y); a.u[7] = pack2(f3.z, f3.w);
    }
#pragma unroll
    for (int dt = 0; dt < 8; ++dt) {
      ABu bm;
      bm.q[0] = *(const uv4*)&wt[dt * 16 + nB][hi * 4];
      bm.q[1] = *(const uv4*)&wt[dt * 16 + nB][hi * 4 + 8];
      acc[dt] = __builtin_amdgcn_wmma_f32_16x16x32_bf16(false, a.v, false, bm.v,
                                                        (short)0, acc[dt],
                                                        false, false);
    }
  }

#pragma unroll
  for (int dt = 0; dt < 8; ++dt)
#pragma unroll
    for (int r = 0; r < 8; ++r) {
      const int m   = r + (hi ? 8 : 0);
      const int row = rowBase + m;
      const int d   = dt * 16 + nB;
      const unsigned short h = f2bf(acc[dt][r]);
      if (MODE == 0) {
        out[(size_t)row * 128 + d] = h;
      } else {                               // V transposed: [b][d][k]
        const int b = row >> 11;
        const int k = row & 2047;
        out[((size_t)b * 128 + d) * 2048 + k] = h;
      }
    }
}

// ---------------------------------------------------------------------------
// Pass 2: flash-style attention. One wave = 16 q-rows; block = 128 q-rows.
// Grid = B * (Sq/128) = 256. K / Vt chunks staged once per block into LDS
// (async global->LDS when available), then consumed by all 8 waves as
// conflict-free ds_load_b128 WMMA operands.
// ---------------------------------------------------------------------------
__global__ __launch_bounds__(256)
void attn_kernel(const unsigned short* __restrict__ Qb,
                 const unsigned short* __restrict__ Kb,
                 const unsigned short* __restrict__ Vt,
                 float* __restrict__ ctx_out, float* __restrict__ w_out) {
  __shared__ __align__(16) unsigned short kt[32][136];       // K chunk, 272B rows
  __shared__ __align__(16) unsigned short vt[128][40];       // Vt chunk, 80B rows
  __shared__ __align__(16) unsigned short p_tile[8][16][40]; // per-wave P tile

  const int lane  = threadIdx.x & 31;
  const int wave  = threadIdx.x >> 5;
  const int b     = blockIdx.x >> 4;
  const int qblk  = blockIdx.x & 15;
  const int qbase = qblk * 128 + wave * 16;
  const int mA = lane & 15;
  const int hi = (lane >> 4) & 1;
  const int nB = lane & 15;
  const float scale = 0.08838834764831845f;  // 1/sqrt(128)

  // Preload Q strip (16x128) as 4 A-operands: two b128 loads each.
  ABu qa[4];
  {
    const unsigned short* qrow =
        Qb + (size_t)(b * 2048 + qbase + mA) * 128 + hi * 16;
#pragma unroll
    for (int c = 0; c < 4; ++c) {
      qa[c].q[0] = *(const uv4*)(qrow + c * 32);
      qa[c].q[1] = *(const uv4*)(qrow + c * 32 + 8);
    }
  }

  const unsigned short* kbat = Kb + (size_t)b * 2048 * 128;
  const unsigned short* vbat = Vt + (size_t)b * 128 * 2048;

  float rm[8], rs[8];
#pragma unroll
  for (int r = 0; r < 8; ++r) { rm[r] = -3.0e38f; rs[r] = 0.0f; }

  // -------- Sweep 1: softmax statistics --------
  for (int tt = 0; tt < 64; ++tt) {
    const int kb2 = tt * 32;
    const unsigned short* ksrc = kbat + (size_t)kb2 * 128;
#pragma unroll
    for (int i = 0; i < 2; ++i) {
      const int o = (threadIdx.x + i * 256) * 8;   // flat elem offset in 32x128
      cp16(ksrc + o, &kt[o >> 7][o & 127]);
    }
    cp16_wait();
    __syncthreads();

#pragma unroll
    for (int h = 0; h < 2; ++h) {
      const unsigned short* krow = &kt[h * 16 + nB][hi * 16];
      v8f s = {};
#pragma unroll
      for (int c = 0; c < 4; ++c) {
        ABu bm;
        bm.q[0] = *(const uv4*)(krow + c * 32);
        bm.q[1] = *(const uv4*)(krow + c * 32 + 8);
        s = __builtin_amdgcn_wmma_f32_16x16x32_bf16(false, qa[c].v, false, bm.v,
                                                    (short)0, s, false, false);
      }
#pragma unroll
      for (int r = 0; r < 8; ++r) {
        const float sv = s[r] * scale;
        const float nm = fmaxf(rm[r], sv);
        rs[r] = rs[r] * __expf(rm[r] - nm) + __expf(sv - nm);
        rm[r] = nm;
      }
    }
    __syncthreads();
  }

  // Combine across the 16-lane half that owns each row group.
#pragma unroll
  for (int r = 0; r < 8; ++r) {
#pragma unroll
    for (int mask = 1; mask < 16; mask <<= 1) {
      const float om = __shfl_xor(rm[r], mask, 32);
      const float os = __shfl_xor(rs[r], mask, 32);
      const float nm = fmaxf(rm[r], om);
      rs[r] = rs[r] * __expf(rm[r] - nm) + os * __expf(om - nm);
      rm[r] = nm;
    }
  }
  float rinv[8];
#pragma unroll
  for (int r = 0; r < 8; ++r) rinv[r] = 1.0f / rs[r];

  v8f ctx[8];
#pragma unroll
  for (int dt = 0; dt < 8; ++dt) ctx[dt] = {};

  // -------- Sweep 2: weights out + context accumulate --------
  for (int tt = 0; tt < 64; ++tt) {
    const int kb2 = tt * 32;
    const unsigned short* ksrc = kbat + (size_t)kb2 * 128;
#pragma unroll
    for (int i = 0; i < 2; ++i) {
      const int o = (threadIdx.x + i * 256) * 8;   // 32x128 K chunk
      cp16(ksrc + o, &kt[o >> 7][o & 127]);
    }
#pragma unroll
    for (int i = 0; i < 2; ++i) {
      const int o = (threadIdx.x + i * 256) * 8;   // 128x32 Vt chunk
      cp16(vbat + (size_t)(o >> 5) * 2048 + kb2 + (o & 31), &vt[o >> 5][o & 31]);
    }
    cp16_wait();
    __syncthreads();

#pragma unroll
    for (int h = 0; h < 2; ++h) {
      const int kb = kb2 + h * 16;
      const unsigned short* krow = &kt[h * 16 + nB][hi * 16];
      v8f s = {};
#pragma unroll
      for (int c = 0; c < 4; ++c) {
        ABu bm;
        bm.q[0] = *(const uv4*)(krow + c * 32);
        bm.q[1] = *(const uv4*)(krow + c * 32 + 8);
        s = __builtin_amdgcn_wmma_f32_16x16x32_bf16(false, qa[c].v, false, bm.v,
                                                    (short)0, s, false, false);
      }
#pragma unroll
      for (int r = 0; r < 8; ++r) {
        const int m = r + (hi ? 8 : 0);
        const float p = __expf(s[r] * scale - rm[r]) * rinv[r];
        w_out[(size_t)(b * 2048 + qbase + m) * 2048 + kb + nB] = p;
        p_tile[wave][m][h * 16 + nB] = f2bf(p);
      }
    }
    asm volatile("s_wait_dscnt 0x0" ::: "memory");  // LDS stores -> crossed loads

    ABu pa;                                         // P (16x32) as A-operand
    {
      const unsigned short* prow = &p_tile[wave][mA][hi * 16];
      pa.q[0] = *(const uv4*)(prow);
      pa.q[1] = *(const uv4*)(prow + 8);
    }
#pragma unroll
    for (int dt = 0; dt < 8; ++dt) {
      const unsigned short* vrow = &vt[dt * 16 + nB][hi * 16];
      ABu bm;
      bm.q[0] = *(const uv4*)(vrow);
      bm.q[1] = *(const uv4*)(vrow + 8);
      ctx[dt] = __builtin_amdgcn_wmma_f32_16x16x32_bf16(false, pa.v, false, bm.v,
                                                        (short)0, ctx[dt],
                                                        false, false);
    }
    __syncthreads();                                // kt/vt/p_tile reuse guard
  }

  // Write context (f32).
#pragma unroll
  for (int dt = 0; dt < 8; ++dt)
#pragma unroll
    for (int r = 0; r < 8; ++r) {
      const int m = r + (hi ? 8 : 0);
      ctx_out[(size_t)(b * 2048 + qbase + m) * 128 + dt * 16 + nB] = ctx[dt][r];
    }
}

// ---------------------------------------------------------------------------
extern "C" void kernel_launch(void* const* d_in, const int* in_sizes, int n_in,
                              void* d_out, int out_size, void* d_ws, size_t ws_size,
                              hipStream_t stream) {
  const float* Lt   = (const float*)d_in[0];   // [16,2048,512]
  const float* rnn  = (const float*)d_in[1];   // [16,2048,512]
  const float* kern = (const float*)d_in[2];   // [2,512,128]
  const float* W    = (const float*)d_in[3];   // [512,128]

  unsigned short* Qb = (unsigned short*)d_ws;                 // [32768][128] bf16
  unsigned short* Kb = Qb + (size_t)16 * 2048 * 128;          // [32768][128] bf16
  unsigned short* Vt = Kb + (size_t)16 * 2048 * 128;          // [16][128][2048] bf16

  float* ctx = (float*)d_out;                                 // [16,2048,128]
  float* wts = ctx + (size_t)16 * 2048 * 128;                 // [16,2048,2048]

  dim3 blk(256), grd(256);
  proj_kernel<0><<<grd, blk, 0, stream>>>(rnn, W, Qb);
  proj_kernel<0><<<grd, blk, 0, stream>>>(Lt, kern, Kb);
  proj_kernel<1><<<grd, blk, 0, stream>>>(Lt, kern + (size_t)512 * 128, Vt);
  attn_kernel<<<grd, blk, 0, stream>>>(Qb, Kb, Vt, ctx, wts);
}
